// HEALDownSampler_46377056863017
// MI455X (gfx1250) — matrix operand order repaired and checked by reading, not compiled, not run on hardware
//
#include <hip/hip_runtime.h>
#include <hip/hip_fp16.h>

#define E_EDGES 196608
#define NRECV   12288
#define BATCH   2

typedef __attribute__((ext_vector_type(16))) _Float16 v16h;
typedef __attribute__((ext_vector_type(8)))  float    v8f;

__device__ __forceinline__ float silu_f(float x) {
    return x / (1.0f + __expf(-x));
}

__device__ __forceinline__ void atomic_add_f32(float* p, float v) {
    __hip_atomic_fetch_add(p, v, __ATOMIC_RELAXED, __HIP_MEMORY_SCOPE_AGENT);
}

// Pack a [K,N] fp32 weight matrix into WMMA B-fragment order (f16):
// layout [N/16][K/32][lane=32][slot=16], element = W[kt*32+8*(lane/16)+(slot&7)+16*(slot/8)][nt*16+lane%16]
__global__ void pack_bfrags_kernel(const float* __restrict__ W,
                                   _Float16* __restrict__ out, int K, int N) {
    int idx = blockIdx.x * blockDim.x + threadIdx.x;
    if (idx >= K * N) return;
    int slot = idx & 15;
    int lane = (idx >> 4) & 31;
    int t    = idx >> 9;
    int nkt  = K >> 5;
    int kt   = t % nkt;
    int nt   = t / nkt;
    out[idx] = (_Float16)W[(kt * 32 + 8 * (lane >> 4) + (slot & 7) + ((slot >> 3) << 4)) * N + nt * 16 + (lane & 15)];
}

// Edge MLP: t = silu(edge_attr @ W1 + b1)  (VALU, K=4)
// h = t @ W2 + b2                          (WMMA f16, K=128)
// then scatter-add h rows into agg_h[recv] (batch-invariant half)
__global__ void __launch_bounds__(256) edge_embed_scatter_kernel(
    const float* __restrict__ edge_attr, const int* __restrict__ recv,
    const float* __restrict__ W1, const float* __restrict__ b1,
    const _Float16* __restrict__ w2f, const float* __restrict__ b2,
    float* __restrict__ agg_h)
{
    __shared__ float sW1[512];
    __shared__ float sb1[128];
    int tid = threadIdx.x;
    for (int i = tid; i < 512; i += 256) sW1[i] = W1[i];
    if (tid < 128) sb1[tid] = b1[tid];
    __syncthreads();

    int lane = tid & 31;
    int wv   = tid >> 5;
    int hi   = lane >> 4;
    int m    = lane & 15;
    int tile = blockIdx.x * 8 + wv;
    int ebase = tile * 16;
    int e     = ebase + m;

    float4 ea = ((const float4*)edge_attr)[e];

    // A fragments: lane's 16 slots all belong to edge row m, K per ISA layout
    v16h afrag[4];
    #pragma unroll
    for (int kt = 0; kt < 4; ++kt) {
        #pragma unroll
        for (int i = 0; i < 16; ++i) {
            int k = kt * 32 + i + 8 * hi + ((i >= 8) ? 8 : 0);
            float s = ea.x * sW1[k] + ea.y * sW1[128 + k]
                    + ea.z * sW1[256 + k] + ea.w * sW1[384 + k] + sb1[k];
            afrag[kt][i] = (_Float16)silu_f(s);
        }
    }

    int rcv[8];
    #pragma unroll
    for (int r = 0; r < 8; ++r) rcv[r] = recv[ebase + 8 * hi + r];

    const v16h* bp = (const v16h*)w2f;
    #pragma unroll
    for (int nt = 0; nt < 8; ++nt) {
        int n = nt * 16 + m;
        float bb = b2[n];
        v8f c;
        #pragma unroll
        for (int r = 0; r < 8; ++r) c[r] = bb;
        #pragma unroll
        for (int kt = 0; kt < 4; ++kt) {
            v16h bf = bp[(nt * 4 + kt) * 32 + lane];
            c = __builtin_amdgcn_wmma_f32_16x16x32_f16(
                    false, afrag[kt], false, bf, (short)0, c, false, false);
        }
        // D layout: lane holds rows r+8*hi, col n -> fused scatter (L2 atomics)
        #pragma unroll
        for (int r = 0; r < 8; ++r)
            atomic_add_f32(&agg_h[(size_t)rcv[r] * 128 + n], c[r]);
    }
}

// Scatter x: one wave per (batch, edge) row; lane handles 4 contiguous floats.
__global__ void scatter_x_kernel(const float* __restrict__ x,
                                 const int* __restrict__ recv,
                                 float* __restrict__ agg_x)
{
    long long gid = (long long)blockIdx.x * 256 + threadIdx.x;
    int lane = (int)(gid & 31);
    long long row = gid >> 5;                 // 0 .. B*E-1
    int b = (int)(row / E_EDGES);
    int e = (int)(row - (long long)b * E_EDGES);
    int node = recv[e];
    float4 v = ((const float4*)(x + row * 128))[lane];
    float* dst = agg_x + ((size_t)b * NRECV + node) * 128 + lane * 4;
    atomic_add_f32(dst + 0, v.x);
    atomic_add_f32(dst + 1, v.y);
    atomic_add_f32(dst + 2, v.z);
    atomic_add_f32(dst + 3, v.w);
}

// Output FFN: out = silu(concat(agg_h, agg_x) @ L1 + c1) @ L2 + c2
// Wave handles one 16-row tile; GEMM1 -> LDS transpose -> GEMM2. All WMMA f16.
__global__ void __launch_bounds__(128) ffn_kernel(
    const float* __restrict__ agg_h, const float* __restrict__ agg_x,
    const _Float16* __restrict__ l1f, const float* __restrict__ c1,
    const _Float16* __restrict__ l2f, const float* __restrict__ c2,
    float* __restrict__ out)
{
    __shared__ _Float16 uLds[4][16][272];   // pad 16 halves -> stride 544B (bank spread)
    int tid  = threadIdx.x;
    int lane = tid & 31;
    int wv   = tid >> 5;
    int hi   = lane >> 4;
    int m    = lane & 15;
    int tile = blockIdx.x * 4 + wv;
    int row  = tile * 16 + m;               // flat over B*NRECV
    int b    = row / NRECV;
    int node = row - b * NRECV;

    const float* ph = agg_h + (size_t)node * 128;
    const float* px = agg_x + ((size_t)b * NRECV + node) * 128 - 128; // indexed by k>=128

    // A fragments for GEMM1: two contiguous 8-float runs per kt (ISA A layout)
    v16h a[8];
    #pragma unroll
    for (int kt = 0; kt < 8; ++kt) {
        const float* src = (kt < 4) ? ph : px;
        int k0 = kt * 32 + 8 * hi;
        #pragma unroll
        for (int i = 0; i < 8; ++i) a[kt][i]     = (_Float16)src[k0 + i];
        #pragma unroll
        for (int i = 0; i < 8; ++i) a[kt][8 + i] = (_Float16)src[k0 + 16 + i];
    }

    const v16h* b1p = (const v16h*)l1f;
    #pragma unroll
    for (int nt = 0; nt < 16; ++nt) {
        float bb = c1[nt * 16 + m];
        v8f c;
        #pragma unroll
        for (int r = 0; r < 8; ++r) c[r] = bb;
        #pragma unroll
        for (int kt = 0; kt < 8; ++kt) {
            v16h bf = b1p[(nt * 8 + kt) * 32 + lane];
            c = __builtin_amdgcn_wmma_f32_16x16x32_f16(
                    false, a[kt], false, bf, (short)0, c, false, false);
        }
        #pragma unroll
        for (int r = 0; r < 8; ++r)
            uLds[wv][r + 8 * hi][nt * 16 + m] = (_Float16)silu_f(c[r]);
    }
    __syncthreads();

    // Rebuild A fragments for GEMM2 from LDS (D-layout -> A-layout transpose)
    #pragma unroll
    for (int kt = 0; kt < 8; ++kt) {
        int k0 = kt * 32 + 8 * hi;
        #pragma unroll
        for (int i = 0; i < 8; ++i) a[kt][i]     = uLds[wv][m][k0 + i];
        #pragma unroll
        for (int i = 0; i < 8; ++i) a[kt][8 + i] = uLds[wv][m][k0 + 16 + i];
    }

    const v16h* b2p = (const v16h*)l2f;
    #pragma unroll
    for (int nt = 0; nt < 16; ++nt) {
        float bb = c2[nt * 16 + m];
        v8f c;
        #pragma unroll
        for (int r = 0; r < 8; ++r) c[r] = bb;
        #pragma unroll
        for (int kt = 0; kt < 8; ++kt) {
            v16h bf = b2p[(nt * 8 + kt) * 32 + lane];
            c = __builtin_amdgcn_wmma_f32_16x16x32_f16(
                    false, a[kt], false, bf, (short)0, c, false, false);
        }
        #pragma unroll
        for (int r = 0; r < 8; ++r)
            out[(size_t)(tile * 16 + r + 8 * hi) * 256 + nt * 16 + m] = c[r];
    }
}

extern "C" void kernel_launch(void* const* d_in, const int* in_sizes, int n_in,
                              void* d_out, int out_size, void* d_ws, size_t ws_size,
                              hipStream_t stream)
{
    const float* x         = (const float*)d_in[0];
    const float* edge_attr = (const float*)d_in[1];
    const int*   edge_idx  = (const int*)d_in[2];
    const float* W1 = (const float*)d_in[3];
    const float* b1 = (const float*)d_in[4];
    const float* W2 = (const float*)d_in[5];
    const float* b2 = (const float*)d_in[6];
    const float* L1 = (const float*)d_in[7];
    const float* c1 = (const float*)d_in[8];
    const float* L2 = (const float*)d_in[9];
    const float* c2 = (const float*)d_in[10];
    float* out = (float*)d_out;
    const int* recv = edge_idx + E_EDGES;   // edge_index[1]

    // Workspace layout (all 256B aligned):
    //   agg_h : NRECV*128 fp32            =  6,291,456 B
    //   agg_x : B*NRECV*128 fp32          = 12,582,912 B
    //   w2f   : 128*128 f16 frag-packed   =     32,768 B
    //   l1f   : 256*256 f16 frag-packed   =    131,072 B
    //   l2f   : 256*256 f16 frag-packed   =    131,072 B
    char* ws = (char*)d_ws;
    float*    agg_h = (float*)ws;
    float*    agg_x = (float*)(ws + 6291456);
    _Float16* w2f   = (_Float16*)(ws + 18874368);
    _Float16* l1f   = (_Float16*)(ws + 18907136);
    _Float16* l2f   = (_Float16*)(ws + 19038208);

    // Zero accumulators every call (graph-capture-safe async memset).
    hipMemsetAsync(ws, 0, 18874368, stream);

    // Pre-pack weights into WMMA B-fragment order (f16).
    pack_bfrags_kernel<<<(128 * 128 + 255) / 256, 256, 0, stream>>>(W2, w2f, 128, 128);
    pack_bfrags_kernel<<<(256 * 256 + 255) / 256, 256, 0, stream>>>(L1, l1f, 256, 256);
    pack_bfrags_kernel<<<(256 * 256 + 255) / 256, 256, 0, stream>>>(L2, l2f, 256, 256);

    // Edge MLP (WMMA) + fused scatter of batch-invariant half.
    edge_embed_scatter_kernel<<<E_EDGES / 128, 256, 0, stream>>>(
        edge_attr, recv, W1, b1, w2f, b2, agg_h);

    // Scatter per-batch node features.
    scatter_x_kernel<<<(int)(((long long)BATCH * E_EDGES * 32) / 256), 256, 0, stream>>>(
        x, recv, agg_x);

    // Output FFN (two WMMA GEMMs per row tile).
    ffn_kernel<<<(BATCH * NRECV / 16) / 4, 128, 0, stream>>>(
        agg_h, agg_x, l1f, c1, l2f, c2, out);
}
